// LARNN_34419867910481
// MI455X (gfx1250) — matrix operand order repaired and checked by reading, not compiled, not run on hardware
//
#include <hip/hip_runtime.h>
#include <math.h>

// ---------------------------------------------------------------------------
// LARNN scan for gfx1250: persistent kernel, incremental V-buffer, f16 WMMA
// with 2x2 register blocking (32x32 per wave) and a unified phase-1 queue.
// ---------------------------------------------------------------------------

typedef __attribute__((ext_vector_type(16))) _Float16 v16h;
typedef __attribute__((ext_vector_type(8)))  float    v8f;

static constexpr int T_  = 128;
static constexpr int B_  = 128;
static constexpr int I_  = 128;
static constexpr int H_  = 512;
static constexpr int W_  = 16;
static constexpr int H4_ = 2048;
static constexpr float INV_SQRT_DK = 0.04419417382415922f; // 1/sqrt(512)

static constexpr int GRID_  = 48;   // 48 WGs * 8 waves = 384 waves == phase-1 groups
static constexpr int BLOCK_ = 256;

// group counts (32x32 output groups)
static constexpr int GRP_V   = (B_ / 32) * (H_ / 32);    // 4 * 16  = 64
static constexpr int GRP_Q   = (B_ / 32) * (H_ / 32);    // 64
static constexpr int GRP_PRE = (B_ / 32) * (H4_ / 32);   // 4 * 64  = 256

// ---- workspace layout (all offsets multiples of 256 bytes) ----
static constexpr size_t OFF_BAR  = 0;                                   // 2 u32 (+pad)
static constexpr size_t OFF_H    = 256;                                 // f32 B*H
static constexpr size_t OFF_C    = OFF_H    + (size_t)B_*H_*4;          // f32 B*H
static constexpr size_t OFF_VBUF = OFF_C    + (size_t)B_*H_*4;          // f32 W*B*H
static constexpr size_t OFF_Q    = OFF_VBUF + (size_t)W_*B_*H_*4;       // f32 B*H
static constexpr size_t OFF_PRE  = OFF_Q    + (size_t)B_*H_*4;          // f32 B*4H
static constexpr size_t OFF_BSUM = OFF_PRE  + (size_t)B_*H4_*4;         // f32 4H
static constexpr size_t OFF_HBF  = OFF_BSUM + (size_t)H4_*4;            // f16 B*H
static constexpr size_t OFF_CBF  = OFF_HBF  + (size_t)B_*H_*2;          // f16 B*H
static constexpr size_t OFF_ABF  = OFF_CBF  + (size_t)B_*H_*2;          // f16 B*H
static constexpr size_t OFF_X16  = OFF_ABF  + (size_t)B_*H_*2;          // f16 T*B*I
static constexpr size_t OFF_WIT  = OFF_X16  + (size_t)T_*B_*I_*2;       // f16 4H x I   (N-major)
static constexpr size_t OFF_WHT  = OFF_WIT  + (size_t)H4_*I_*2;         // f16 4H x H
static constexpr size_t OFF_WAT  = OFF_WHT  + (size_t)H4_*H_*2;         // f16 4H x H
static constexpr size_t OFF_WVT  = OFF_WAT  + (size_t)H4_*H_*2;         // f16 H x H
static constexpr size_t OFF_WQXT = OFF_WVT  + (size_t)H_*H_*2;          // f16 H x I
static constexpr size_t OFF_WQHT = OFF_WQXT + (size_t)H_*I_*2;          // f16 H x H

union AFrag { v16h v; uint4 q[2]; };
union BFrag { v16h v; uint4 q[2]; };

// ---------------------------------------------------------------------------
// Device-wide sense barrier (agent-scope atomics in d_ws)
// ---------------------------------------------------------------------------
__device__ __forceinline__ void grid_barrier(unsigned* cnt, unsigned* gen, unsigned nwg)
{
    __threadfence();
    __syncthreads();
    if (threadIdx.x == 0) {
        unsigned g = __hip_atomic_load(gen, __ATOMIC_RELAXED, __HIP_MEMORY_SCOPE_AGENT);
        unsigned a = __hip_atomic_fetch_add(cnt, 1u, __ATOMIC_ACQ_REL, __HIP_MEMORY_SCOPE_AGENT);
        if (a == nwg - 1u) {
            __hip_atomic_store(cnt, 0u, __ATOMIC_RELAXED, __HIP_MEMORY_SCOPE_AGENT);
            __hip_atomic_fetch_add(gen, 1u, __ATOMIC_ACQ_REL, __HIP_MEMORY_SCOPE_AGENT);
        } else {
            while (__hip_atomic_load(gen, __ATOMIC_ACQUIRE, __HIP_MEMORY_SCOPE_AGENT) == g) {
                __builtin_amdgcn_s_sleep(8);
            }
        }
    }
    __syncthreads();
}

// ---------------------------------------------------------------------------
// One 32x32 output group (2x2 tiles of 16x16), K stepped by 32 per WMMA.
//   C[M,N] f32  +=/=  bias + A0[M,K0]*B0T' + A1[M,K1]*B1T'
// A: row-major f16 (lda=K). BT: N x K row-major f16 (pre-transposed weight).
// A frag: lane(0-15)=row r, K {8h..8h+7, 16+8h..+7}  -> two b128 loads.
// B frag: lane col n=lane&15, K {8h..8h+7, 16+8h..+7}  -> two b128 loads.
// C/D:    lane col n, VGPR v -> row 8h+v.
// 8 x b128 loads feed 4 x v_wmma (A frags reused over 2 cols, B over 2 rows).
// ---------------------------------------------------------------------------
__device__ __forceinline__ void gemm_group_2x2(
    const _Float16* __restrict__ A0, int lda0, const _Float16* __restrict__ BT0, int K0,
    const _Float16* __restrict__ A1, int lda1, const _Float16* __restrict__ BT1, int K1,
    const float* __restrict__ bias, float* __restrict__ C, int ldc,
    int TN32, int grp, bool accum)
{
    const int lane = threadIdx.x & 31;
    const int half = lane >> 4;
    const int r    = lane & 15;
    const int gm   = grp / TN32;
    const int gn   = grp - gm * TN32;
    const int rbase = gm * 32;
    const int cbase = gn * 32;

    v8f acc00, acc01, acc10, acc11;
    if (accum) {
#pragma unroll
        for (int v = 0; v < 8; ++v) {
            const size_t r0 = (size_t)(rbase + half*8 + v) * ldc;
            const size_t r1 = (size_t)(rbase + 16 + half*8 + v) * ldc;
            acc00[v] = C[r0 + cbase + r];      acc01[v] = C[r0 + cbase + 16 + r];
            acc10[v] = C[r1 + cbase + r];      acc11[v] = C[r1 + cbase + 16 + r];
        }
    } else {
        const float b0 = bias ? bias[cbase + r]      : 0.0f;
        const float b1 = bias ? bias[cbase + 16 + r] : 0.0f;
#pragma unroll
        for (int v = 0; v < 8; ++v) { acc00[v] = b0; acc01[v] = b1; acc10[v] = b0; acc11[v] = b1; }
    }

    {
        const _Float16* a0p = A0 + (size_t)(rbase + r) * lda0 + half * 8;
        const _Float16* a1p = a0p + (size_t)16 * lda0;
        const _Float16* b0p = BT0 + (size_t)(cbase + r) * K0 + half * 8;
        const _Float16* b1p = b0p + (size_t)16 * K0;
        for (int k = 0; k < K0; k += 32) {
            AFrag a0, a1; BFrag b0, b1;
            a0.q[0] = *(const uint4*)(a0p + k);      a0.q[1] = *(const uint4*)(a0p + k + 16);
            a1.q[0] = *(const uint4*)(a1p + k);      a1.q[1] = *(const uint4*)(a1p + k + 16);
            b0.q[0] = *(const uint4*)(b0p + k);      b0.q[1] = *(const uint4*)(b0p + k + 16);
            b1.q[0] = *(const uint4*)(b1p + k);      b1.q[1] = *(const uint4*)(b1p + k + 16);
            acc00 = __builtin_amdgcn_wmma_f32_16x16x32_f16(false, a0.v, false, b0.v, (short)0, acc00, false, false);
            acc01 = __builtin_amdgcn_wmma_f32_16x16x32_f16(false, a0.v, false, b1.v, (short)0, acc01, false, false);
            acc10 = __builtin_amdgcn_wmma_f32_16x16x32_f16(false, a1.v, false, b0.v, (short)0, acc10, false, false);
            acc11 = __builtin_amdgcn_wmma_f32_16x16x32_f16(false, a1.v, false, b1.v, (short)0, acc11, false, false);
        }
    }
    if (A1) {
        const _Float16* a0p = A1 + (size_t)(rbase + r) * lda1 + half * 8;
        const _Float16* a1p = a0p + (size_t)16 * lda1;
        const _Float16* b0p = BT1 + (size_t)(cbase + r) * K1 + half * 8;
        const _Float16* b1p = b0p + (size_t)16 * K1;
        for (int k = 0; k < K1; k += 32) {
            AFrag a0, a1; BFrag b0, b1;
            a0.q[0] = *(const uint4*)(a0p + k);      a0.q[1] = *(const uint4*)(a0p + k + 16);
            a1.q[0] = *(const uint4*)(a1p + k);      a1.q[1] = *(const uint4*)(a1p + k + 16);
            b0.q[0] = *(const uint4*)(b0p + k);      b0.q[1] = *(const uint4*)(b0p + k + 16);
            b1.q[0] = *(const uint4*)(b1p + k);      b1.q[1] = *(const uint4*)(b1p + k + 16);
            acc00 = __builtin_amdgcn_wmma_f32_16x16x32_f16(false, a0.v, false, b0.v, (short)0, acc00, false, false);
            acc01 = __builtin_amdgcn_wmma_f32_16x16x32_f16(false, a0.v, false, b1.v, (short)0, acc01, false, false);
            acc10 = __builtin_amdgcn_wmma_f32_16x16x32_f16(false, a1.v, false, b0.v, (short)0, acc10, false, false);
            acc11 = __builtin_amdgcn_wmma_f32_16x16x32_f16(false, a1.v, false, b1.v, (short)0, acc11, false, false);
        }
    }

#pragma unroll
    for (int v = 0; v < 8; ++v) {
        const size_t r0 = (size_t)(rbase + half*8 + v) * ldc;
        const size_t r1 = (size_t)(rbase + 16 + half*8 + v) * ldc;
        C[r0 + cbase + r]      = acc00[v];
        C[r0 + cbase + 16 + r] = acc01[v];
        C[r1 + cbase + r]      = acc10[v];
        C[r1 + cbase + 16 + r] = acc11[v];
    }
}

// ---------------------------------------------------------------------------
// Prologue: convert/transpose weights to f16, convert x, init state/barrier.
// ---------------------------------------------------------------------------
__global__ void larnn_prologue(const float* __restrict__ x,  const float* __restrict__ Wi,
                               const float* __restrict__ bi, const float* __restrict__ Wh,
                               const float* __restrict__ Wv, const float* __restrict__ bv,
                               const float* __restrict__ Wq, const float* __restrict__ bq,
                               const float* __restrict__ Wa, const float* __restrict__ ba,
                               char* __restrict__ ws)
{
    const size_t tid    = (size_t)blockIdx.x * blockDim.x + threadIdx.x;
    const size_t stride = (size_t)gridDim.x * blockDim.x;

    unsigned*  bar  = (unsigned*)(ws + OFF_BAR);
    float*     h    = (float*)(ws + OFF_H);
    float*     c    = (float*)(ws + OFF_C);
    float*     Vbuf = (float*)(ws + OFF_VBUF);
    float*     bsum = (float*)(ws + OFF_BSUM);
    _Float16*  hbf  = (_Float16*)(ws + OFF_HBF);
    _Float16*  cbf  = (_Float16*)(ws + OFF_CBF);
    _Float16*  x16  = (_Float16*)(ws + OFF_X16);
    _Float16*  WiT  = (_Float16*)(ws + OFF_WIT);
    _Float16*  WhT  = (_Float16*)(ws + OFF_WHT);
    _Float16*  WaT  = (_Float16*)(ws + OFF_WAT);
    _Float16*  WvT  = (_Float16*)(ws + OFF_WVT);
    _Float16*  WqXT = (_Float16*)(ws + OFF_WQXT);
    _Float16*  WqHT = (_Float16*)(ws + OFF_WQHT);

    if (tid < 64) bar[tid] = 0u;                                       // barrier state

    for (size_t i = tid; i < (size_t)T_*B_*I_; i += stride) x16[i] = (_Float16)x[i];

    for (size_t i = tid; i < (size_t)H4_*I_; i += stride) {            // WiT[n][k] = Wi[k][n]
        size_t n = i / I_, k = i % I_;
        WiT[i] = (_Float16)Wi[k*H4_ + n];
    }
    for (size_t i = tid; i < (size_t)H4_*H_; i += stride) {            // WhT / WaT
        size_t n = i / H_, k = i % H_;
        WhT[i] = (_Float16)Wh[k*H4_ + n];
        WaT[i] = (_Float16)Wa[k*H4_ + n];
    }
    for (size_t i = tid; i < (size_t)H_*H_; i += stride) {             // WvT, WqHT
        size_t n = i / H_, k = i % H_;
        WvT[i]  = (_Float16)Wv[k*H_ + n];
        WqHT[i] = (_Float16)Wq[(I_ + k)*H_ + n];
    }
    for (size_t i = tid; i < (size_t)H_*I_; i += stride) {             // WqXT
        size_t n = i / I_, k = i % I_;
        WqXT[i] = (_Float16)Wq[k*H_ + n];
    }
    for (size_t i = tid; i < (size_t)B_*H_; i += stride) {             // zero state
        h[i] = 0.0f; c[i] = 0.0f;
        hbf[i] = (_Float16)0.0f; cbf[i] = (_Float16)0.0f;
    }
    for (size_t i = tid; i < (size_t)W_*B_*H_; i += stride)            // Vbuf = bv (== 0 @ Wv + bv)
        Vbuf[i] = bv[i % H_];
    for (size_t i = tid; i < (size_t)H4_; i += stride)                 // fused bias bi+ba
        bsum[i] = bi[i] + ba[i];
}

// ---------------------------------------------------------------------------
// Persistent scan kernel: 128 steps, 4 grid barriers per step.
// ---------------------------------------------------------------------------
__global__ __launch_bounds__(BLOCK_, 1)
void larnn_scan(const float* __restrict__ bv, const float* __restrict__ bq,
                char* __restrict__ ws, float* __restrict__ out)
{
    unsigned*       cnt  = (unsigned*)(ws + OFF_BAR);
    unsigned*       gen  = cnt + 1;
    float*          h    = (float*)(ws + OFF_H);
    float*          c    = (float*)(ws + OFF_C);
    float*          Vbuf = (float*)(ws + OFF_VBUF);
    float*          Qb   = (float*)(ws + OFF_Q);
    float*          pre  = (float*)(ws + OFF_PRE);
    const float*    bsum = (const float*)(ws + OFF_BSUM);
    _Float16*       hbf  = (_Float16*)(ws + OFF_HBF);
    _Float16*       cbf  = (_Float16*)(ws + OFF_CBF);
    _Float16*       abf  = (_Float16*)(ws + OFF_ABF);
    const _Float16* x16  = (const _Float16*)(ws + OFF_X16);
    const _Float16* WiT  = (const _Float16*)(ws + OFF_WIT);
    const _Float16* WhT  = (const _Float16*)(ws + OFF_WHT);
    const _Float16* WaT  = (const _Float16*)(ws + OFF_WAT);
    const _Float16* WvT  = (const _Float16*)(ws + OFF_WVT);
    const _Float16* WqXT = (const _Float16*)(ws + OFF_WQXT);
    const _Float16* WqHT = (const _Float16*)(ws + OFF_WQHT);

    const int lane = threadIdx.x & 31;
    const int wpb  = blockDim.x >> 5;
    const int gw   = blockIdx.x * wpb + (threadIdx.x >> 5);
    const int NW   = gridDim.x * wpb;
    const unsigned nwg = gridDim.x;
    const size_t gtid    = (size_t)blockIdx.x * blockDim.x + threadIdx.x;
    const size_t gstride = (size_t)gridDim.x * blockDim.x;

    for (int t = 0; t < T_; ++t) {
        const _Float16* xt = x16 + (size_t)t * B_ * I_;

        // ---- Phase 1: unified queue of 32x32 GEMM groups -------------------
        // [0,GV): Vbuf[t%W] = c_{t-1} @ Wv + bv      (skipped at t==0)
        // [GV,GV+64): Q = x_t @ Wq[:I] + h @ Wq[I:] + bq
        // [GV+64,GV+320): preact = x_t @ Wi + h @ Wh + (bi+ba)
        const int GV    = (t > 0) ? GRP_V : 0;
        const int total = GV + GRP_Q + GRP_PRE;
        float* Vslot = Vbuf + (size_t)(t % W_) * B_ * H_;
        for (int g = gw; g < total; g += NW) {
            if (g < GV) {
                gemm_group_2x2(cbf, H_, WvT, H_, nullptr, 0, nullptr, 0,
                               bv, Vslot, H_, H_ / 32, g, false);
            } else if (g < GV + GRP_Q) {
                gemm_group_2x2(xt, I_, WqXT, I_, hbf, H_, WqHT, H_,
                               bq, Qb, H_, H_ / 32, g - GV, false);
            } else {
                gemm_group_2x2(xt, I_, WiT, I_, hbf, H_, WhT, H_,
                               bsum, pre, H4_, H4_ / 32, g - GV - GRP_Q, false);
            }
        }
        grid_barrier(cnt, gen, nwg);

        // ---- Phase 2: W=16-way attention (VALU, one wave per batch row) ----
        const int wmax = (t + 1 < W_) ? (t + 1) : W_;
        for (int b = gw; b < B_; b += NW) {
            const float* Q = Qb + (size_t)b * H_;
            float qreg[16];
#pragma unroll
            for (int i = 0; i < 16; ++i) qreg[i] = Q[lane + 32 * i];

            float s[W_];
#pragma unroll
            for (int w = 0; w < W_; ++w) {
                const float* V = Vbuf + ((size_t)w * B_ + b) * H_;
                float p = 0.0f;
#pragma unroll
                for (int i = 0; i < 16; ++i) p += qreg[i] * V[lane + 32 * i];
#pragma unroll
                for (int off = 16; off > 0; off >>= 1) p += __shfl_xor(p, off, 32);
                s[w] = (w < wmax) ? p * INV_SQRT_DK : -1e30f;
            }
            float m = s[0];
#pragma unroll
            for (int w = 1; w < W_; ++w) m = fmaxf(m, s[w]);
            float sum = 0.0f;
#pragma unroll
            for (int w = 0; w < W_; ++w) { s[w] = __expf(s[w] - m); sum += s[w]; }
            const float inv = 1.0f / sum;
#pragma unroll
            for (int i = 0; i < 16; ++i) {
                const int hh = lane + 32 * i;
                float a = 0.0f;
#pragma unroll
                for (int w = 0; w < W_; ++w) a += s[w] * Vbuf[((size_t)w * B_ + b) * H_ + hh];
                abf[(size_t)b * H_ + hh] = (_Float16)(a * inv);
            }
        }
        grid_barrier(cnt, gen, nwg);

        // ---- Phase 3a: preact += attn @ Wa ---------------------------------
        for (int g = gw; g < GRP_PRE; g += NW) {
            gemm_group_2x2(abf, H_, WaT, H_, nullptr, 0, nullptr, 0,
                           nullptr, pre, H4_, H4_ / 32, g, true);
        }
        grid_barrier(cnt, gen, nwg);

        // ---- Phase 3b: gates, state update, output -------------------------
        for (size_t j = gtid; j < (size_t)B_ * H_; j += gstride) {
            const size_t b = j >> 9, n = j & (H_ - 1);
            const float* pr = pre + b * H4_;
            const float i_t = 1.0f / (1.0f + __expf(-pr[n]));
            const float f_t = 1.0f / (1.0f + __expf(-pr[H_ + n]));
            const float o_t = 1.0f / (1.0f + __expf(-pr[2 * H_ + n]));
            const float g_t = tanhf(pr[3 * H_ + n]);
            const float cn  = c[j] * f_t + i_t * g_t;
            const float hn  = o_t * tanhf(cn);
            c[j] = cn; h[j] = hn;
            cbf[j] = (_Float16)cn; hbf[j] = (_Float16)hn;
            out[(size_t)t * B_ * H_ + j] = hn;
        }
        grid_barrier(cnt, gen, nwg);
    }
}

// ---------------------------------------------------------------------------
extern "C" void kernel_launch(void* const* d_in, const int* in_sizes, int n_in,
                              void* d_out, int out_size, void* d_ws, size_t ws_size,
                              hipStream_t stream)
{
    const float* x  = (const float*)d_in[0];
    const float* Wi = (const float*)d_in[1];
    const float* bi = (const float*)d_in[2];
    const float* Wh = (const float*)d_in[3];
    const float* Wv = (const float*)d_in[4];
    const float* bv = (const float*)d_in[5];
    const float* Wq = (const float*)d_in[6];
    const float* bq = (const float*)d_in[7];
    const float* Wa = (const float*)d_in[8];
    const float* ba = (const float*)d_in[9];
    char* ws = (char*)d_ws;

    larnn_prologue<<<256, BLOCK_, 0, stream>>>(x, Wi, bi, Wh, Wv, bv, Wq, bq, Wa, ba, ws);
    larnn_scan<<<GRID_, BLOCK_, 0, stream>>>(bv, bq, ws, (float*)d_out);
}
